// EnhancedGraphTransformer_14912126452018
// MI455X (gfx1250) — compile-verified, matrix operand order checked by reading, not run on hardware
//
#include <hip/hip_runtime.h>
#include <hip/hip_bf16.h>

typedef __attribute__((ext_vector_type(16))) _Float16 v16h;
typedef __attribute__((ext_vector_type(8)))  _Float16 h8;
typedef __attribute__((ext_vector_type(4)))  _Float16 h4;
typedef __attribute__((ext_vector_type(8)))  float    v8f;

#define IN_DIM 64
#define H_DIM  128
#define NHEAD  8
#define HD_DIM 128
#define HC_DIM 1024   // NHEAD * HD_DIM
#define FF_DIM 512
#define LAYERS 4

union V16U { v16h v; struct { h8 lo; h8 hi; } p; };

// ---------- helpers: ordered-uint encoding for float atomic max ----------
__device__ __forceinline__ unsigned f32_ord(float f) {
  unsigned u = __float_as_uint(f);
  return (u & 0x80000000u) ? ~u : (u | 0x80000000u);
}
__device__ __forceinline__ float ord_f32(unsigned e) {
  unsigned u = (e & 0x80000000u) ? (e & 0x7FFFFFFFu) : ~e;
  return __uint_as_float(u);
}

// ---------------------------------------------------------------------------
// Weight pack: W[K,Nc] f32 row-major  ->  fragment-ordered f16.
// Record (kb32, ch) holds a 32x16 K-major tile laid out so that lane l's
// 16 fragment halves are contiguous:  Wp[(((kb32*nchunks)+ch)*32 + l)*16 + e]
// where e covers K = kb32*32 + (l>=16?8:0) + (e>=8?16:0) + (e&7), col = ch*16 + (l&15).
// ---------------------------------------------------------------------------
__global__ void pack_w_kernel(const float* __restrict__ W, _Float16* __restrict__ Wp,
                              int K, int Nc) {
  int idx = blockIdx.x * blockDim.x + threadIdx.x;
  int total = (K >> 5) * (Nc >> 4) * 512;
  if (idx >= total) return;
  int e    = idx & 15;
  int lane = (idx >> 4) & 31;
  int t    = idx >> 9;                 // kb32 * nchunks + ch
  int nchunks = Nc >> 4;
  int kb32 = t / nchunks;
  int ch   = t - kb32 * nchunks;
  int col  = (ch << 4) + (lane & 15);
  int kk   = (kb32 << 5) + ((lane >> 4) << 3) + ((e >> 3) << 4) + (e & 7);
  Wp[idx] = (_Float16)W[(size_t)kk * Nc + col];
}

// ---------------------------------------------------------------------------
// WMMA GEMM: C[M,Nc] = A[M,K](f32) * W(packed f16) + bias, optional ReLU.
// One block = 16 output rows staged to LDS as f16; 8 waves sweep 16-col chunks.
// K templated (64/128/512) -> fully unrolled chain of v_wmma_f32_16x16x32_f16.
// Requires M % 16 == 0 (true here: N=20000).
// ---------------------------------------------------------------------------
template<int K, int RELU>
__global__ __launch_bounds__(256) void wmma_gemm_kernel(
    const float* __restrict__ A, const _Float16* __restrict__ Wp,
    const float* __restrict__ bias, float* __restrict__ C, int Nc)
{
  __shared__ _Float16 As[16 * K];
  const int row0 = blockIdx.x * 16;
  const int tid  = threadIdx.x;

  // stage contiguous 16xK A tile (f32 -> f16) into LDS, vectorized
  const float4* src4 = (const float4*)(A + (size_t)row0 * K);
  h4* dst4 = (h4*)As;
  for (int i = tid; i < (16 * K) / 4; i += 256) {
    float4 v = src4[i];
    h4 hv; hv.x = (_Float16)v.x; hv.y = (_Float16)v.y;
    hv.z = (_Float16)v.z; hv.w = (_Float16)v.w;
    dst4[i] = hv;
  }
  __syncthreads();

  const int wave = tid >> 5;
  const int lane = tid & 31;
  const int r    = lane & 15;          // A row within tile / B column within chunk
  const int koff = (lane >> 4) << 3;   // 0 or 8
  const int hi8  = (lane >> 4) << 3;   // C/D row offset for lanes 16-31
  const int nchunks = Nc >> 4;
  constexpr int NKB = K / 32;

  for (int ch = wave; ch < nchunks; ch += 8) {
    // prefetch next chunk's packed weights (global_prefetch_b8)
    if (ch + 8 < nchunks)
      __builtin_prefetch(Wp + ((size_t)(ch + 8) * 32 + lane) * 16, 0, 3);

    v8f acc = {};
#pragma unroll
    for (int kb32 = 0; kb32 < NKB; ++kb32) {
      const int kb = kb32 * 32;
      V16U af, bf;
      af.p.lo = *(const h8*)(As + r * K + kb + koff);
      af.p.hi = *(const h8*)(As + r * K + kb + koff + 16);
      bf.v = *(const v16h*)(Wp + (((size_t)kb32 * nchunks + ch) * 32 + lane) * 16);
      acc = __builtin_amdgcn_wmma_f32_16x16x32_f16(
          false, af.v, false, bf.v, (short)0, acc, false, false);
    }

    const int col = (ch << 4) + r;
    const float bcol = bias[col];
    float* cp = C + (size_t)(row0 + hi8) * Nc + col;
#pragma unroll
    for (int i = 0; i < 8; ++i) {
      float v = acc[i] + bcol;
      if (RELU) v = fmaxf(v, 0.0f);
      cp[(size_t)i * Nc] = v;
    }
  }
}

// ---------------------------------------------------------------------------
// small utility kernels
// ---------------------------------------------------------------------------
__global__ void fill0_kernel(unsigned* __restrict__ p, int n) {
  int i = blockIdx.x * blockDim.x + threadIdx.x;
  if (i < n) p[i] = 0u;
}
__global__ void copy_kernel(float* __restrict__ d, const float* __restrict__ s, int n) {
  int i = blockIdx.x * blockDim.x + threadIdx.x;
  if (i < n) d[i] = s[i];
}
__global__ void add_kernel(float* __restrict__ d, const float* __restrict__ s, int n) {
  int i = blockIdx.x * blockDim.x + threadIdx.x;
  if (i < n) d[i] += s[i];
}

// ---------------------------------------------------------------------------
// edge attention kernels
// ---------------------------------------------------------------------------
__global__ void edge_alpha_kernel(const float* __restrict__ q, const float* __restrict__ k,
                                  const int* __restrict__ src, const int* __restrict__ dst,
                                  float* __restrict__ alpha, unsigned* __restrict__ mmax, int E)
{
  int idx = blockIdx.x * blockDim.x + threadIdx.x;
  if (idx >= E * NHEAD) return;
  int e = idx >> 3, nh = idx & 7;
  const float4* qp = (const float4*)(q + (size_t)dst[e] * HC_DIM + nh * HD_DIM);
  const float4* kp = (const float4*)(k + (size_t)src[e] * HC_DIM + nh * HD_DIM);
  float s = 0.0f;
#pragma unroll 4
  for (int j = 0; j < HD_DIM / 4; ++j) {
    float4 a = qp[j], b = kp[j];
    s += a.x * b.x + a.y * b.y + a.z * b.z + a.w * b.w;
  }
  s *= 0.08838834764831845f;   // 1/sqrt(128)
  alpha[idx] = s;
  atomicMax(&mmax[(size_t)dst[e] * NHEAD + nh], f32_ord(s));
}

__global__ void edge_exp_kernel(float* __restrict__ alpha, const unsigned* __restrict__ mmax,
                                float* __restrict__ denom, const int* __restrict__ dst, int E)
{
  int idx = blockIdx.x * blockDim.x + threadIdx.x;
  if (idx >= E * NHEAD) return;
  int e = idx >> 3, nh = idx & 7;
  float m  = ord_f32(mmax[(size_t)dst[e] * NHEAD + nh]);
  float ea = __expf(alpha[idx] - m);
  alpha[idx] = ea;
  atomicAdd(&denom[(size_t)dst[e] * NHEAD + nh], ea);
}

__global__ void edge_agg_kernel(const float* __restrict__ alpha, const float* __restrict__ denom,
                                const float* __restrict__ v, float* __restrict__ agg,
                                const int* __restrict__ src, const int* __restrict__ dst, int E)
{
  int idx = blockIdx.x * blockDim.x + threadIdx.x;   // over E * NHEAD * (HD/4)
  if (idx >= E * NHEAD * (HD_DIM / 4)) return;
  int e   = idx >> 8;           // NHEAD * HD/4 = 256
  int rem = idx & 255;
  int nh  = rem >> 5;
  int g   = rem & 31;
  float w = alpha[(size_t)e * NHEAD + nh] / denom[(size_t)dst[e] * NHEAD + nh];
  const float4 vv = *(const float4*)(v + (size_t)src[e] * HC_DIM + nh * HD_DIM + g * 4);
  float* ap = agg + (size_t)dst[e] * HC_DIM + nh * HD_DIM + g * 4;
  atomicAdd(ap + 0, vv.x * w);
  atomicAdd(ap + 1, vv.y * w);
  atomicAdd(ap + 2, vv.z * w);
  atomicAdd(ap + 3, vv.w * w);
}

// a[n,d] = mean_heads(agg[n,:,d]) + root[n,d]   (in place into root)
__global__ void head_mean_kernel(const float* __restrict__ agg, float* __restrict__ root, int N)
{
  int idx = blockIdx.x * blockDim.x + threadIdx.x;
  if (idx >= N * H_DIM) return;
  int n = idx >> 7, d = idx & 127;
  float s = 0.0f;
#pragma unroll
  for (int nh = 0; nh < NHEAD; ++nh) s += agg[(size_t)n * HC_DIM + nh * HD_DIM + d];
  root[idx] += s * (1.0f / NHEAD);
}

// h = LayerNorm(h + a) * s + b  — one wave (32 lanes) per row, H=128
__global__ __launch_bounds__(256) void ln_residual_kernel(
    float* __restrict__ h, const float* __restrict__ a,
    const float* __restrict__ sc, const float* __restrict__ bi, int M)
{
  int row  = blockIdx.x * (blockDim.x >> 5) + (threadIdx.x >> 5);
  int lane = threadIdx.x & 31;
  if (row >= M) return;
  float x[4];
  float sum = 0.0f;
#pragma unroll
  for (int j = 0; j < 4; ++j) {
    int c = lane * 4 + j;
    x[j] = h[(size_t)row * H_DIM + c] + a[(size_t)row * H_DIM + c];
    sum += x[j];
  }
  for (int off = 16; off > 0; off >>= 1) sum += __shfl_xor(sum, off, 32);
  float mean = sum * (1.0f / H_DIM);
  float var = 0.0f;
#pragma unroll
  for (int j = 0; j < 4; ++j) { float d = x[j] - mean; var += d * d; }
  for (int off = 16; off > 0; off >>= 1) var += __shfl_xor(var, off, 32);
  float inv = rsqrtf(var * (1.0f / H_DIM) + 1e-5f);
#pragma unroll
  for (int j = 0; j < 4; ++j) {
    int c = lane * 4 + j;
    h[(size_t)row * H_DIM + c] = (x[j] - mean) * inv * sc[c] + bi[c];
  }
}

// ---------------------------------------------------------------------------
// pooling + output head
// ---------------------------------------------------------------------------
__global__ void pool_cnt_kernel(const int* __restrict__ batch, float* __restrict__ cnt, int N) {
  int n = blockIdx.x * blockDim.x + threadIdx.x;
  if (n < N) atomicAdd(&cnt[batch[n]], 1.0f);
}
__global__ void pool_acc_kernel(const float* __restrict__ h, const int* __restrict__ batch,
                                float* __restrict__ pool, int N) {
  int idx = blockIdx.x * blockDim.x + threadIdx.x;
  if (idx >= N * H_DIM) return;
  int n = idx >> 7, d = idx & 127;
  atomicAdd(&pool[(size_t)batch[n] * H_DIM + d], h[idx]);
}
__global__ void pool_final_kernel(const float* __restrict__ pool, const float* __restrict__ cnt,
                                  float* __restrict__ gr, int B) {
  int idx = blockIdx.x * blockDim.x + threadIdx.x;
  if (idx >= B * H_DIM) return;
  int g = idx >> 7;
  gr[idx] = pool[idx] / fmaxf(cnt[g], 1.0f);
}
__global__ void head_kernel(const float* __restrict__ gr, const float* __restrict__ Wo1,
                            const float* __restrict__ bo1, const float* __restrict__ Wo2,
                            const float* __restrict__ bo2, float* __restrict__ out, int B) {
  int g = blockIdx.x * blockDim.x + threadIdx.x;
  if (g >= B) return;
  float o = bo2[0];
  for (int j = 0; j < H_DIM / 2; ++j) {
    float a = bo1[j];
    for (int d = 0; d < H_DIM; ++d) a += gr[(size_t)g * H_DIM + d] * Wo1[d * (H_DIM / 2) + j];
    o += fmaxf(a, 0.0f) * Wo2[j];
  }
  out[g] = o;
}

// ---------------------------------------------------------------------------
// host orchestration
// ---------------------------------------------------------------------------
static inline int ceil_div(long long a, int b) { return (int)((a + b - 1) / b); }

extern "C" void kernel_launch(void* const* d_in, const int* in_sizes, int n_in,
                              void* d_out, int out_size, void* d_ws, size_t ws_size,
                              hipStream_t stream) {
  const float* x    = (const float*)d_in[0];
  const int*   ei   = (const int*)  d_in[1];
  const int*   batch= (const int*)  d_in[2];
  const float* Win  = (const float*)d_in[3];
  const float* b_in = (const float*)d_in[4];
  const float* Wq   = (const float*)d_in[5];
  const float* bq   = (const float*)d_in[6];
  const float* Wk   = (const float*)d_in[7];
  const float* bk   = (const float*)d_in[8];
  const float* Wv   = (const float*)d_in[9];
  const float* bv   = (const float*)d_in[10];
  const float* Wsr  = (const float*)d_in[11];
  const float* bs   = (const float*)d_in[12];
  const float* ln1s = (const float*)d_in[13];
  const float* ln1b = (const float*)d_in[14];
  const float* ln2s = (const float*)d_in[15];
  const float* ln2b = (const float*)d_in[16];
  const float* W1   = (const float*)d_in[17];
  const float* b1   = (const float*)d_in[18];
  const float* W2   = (const float*)d_in[19];
  const float* b2   = (const float*)d_in[20];
  const float* Wsp  = (const float*)d_in[21];
  const float* bsp  = (const float*)d_in[22];
  const float* Wo1  = (const float*)d_in[23];
  const float* bo1  = (const float*)d_in[24];
  const float* Wo2  = (const float*)d_in[25];
  const float* bo2  = (const float*)d_in[26];

  const int N = in_sizes[0] / IN_DIM;   // 20000, multiple of 16
  const int E = in_sizes[1] / 2;
  const int B = out_size / (1 + H_DIM);
  const int* srcI = ei;
  const int* dstI = ei + E;

  // ---- workspace carve-out ----
  char* ws = (char*)d_ws;
  size_t off = 0;
  auto alloc = [&](size_t bytes) -> void* {
    void* p = ws + off;
    off = (off + bytes + 255) & ~(size_t)255;
    return p;
  };
  float*    Hbuf  = (float*)alloc((size_t)N * H_DIM * 4);
  float*    SAV0  = (float*)alloc((size_t)N * H_DIM * 4);
  float*    SAV1  = (float*)alloc((size_t)N * H_DIM * 4);
  float*    Qbuf  = (float*)alloc((size_t)N * HC_DIM * 4);   // also V, also FFN hidden
  float*    Kbuf  = (float*)alloc((size_t)N * HC_DIM * 4);   // also agg accumulator
  float*    ALPHA = (float*)alloc((size_t)E * NHEAD * 4);
  unsigned* MM    = (unsigned*)alloc((size_t)N * NHEAD * 4);
  float*    DEN   = (float*)alloc((size_t)N * NHEAD * 4);
  float*    AR    = (float*)alloc((size_t)N * H_DIM * 4);    // root / ffn out / skip tmp
  float*    POOL  = (float*)alloc((size_t)B * H_DIM * 4);
  float*    CNT   = (float*)alloc((size_t)B * 4);
  _Float16* Win_h = (_Float16*)alloc((size_t)IN_DIM * H_DIM * 2);
  _Float16* Wq_h  = (_Float16*)alloc((size_t)LAYERS * H_DIM * HC_DIM * 2);
  _Float16* Wk_h  = (_Float16*)alloc((size_t)LAYERS * H_DIM * HC_DIM * 2);
  _Float16* Wv_h  = (_Float16*)alloc((size_t)LAYERS * H_DIM * HC_DIM * 2);
  _Float16* Ws_h  = (_Float16*)alloc((size_t)LAYERS * H_DIM * H_DIM * 2);
  _Float16* W1_h  = (_Float16*)alloc((size_t)LAYERS * H_DIM * FF_DIM * 2);
  _Float16* W2_h  = (_Float16*)alloc((size_t)LAYERS * FF_DIM * H_DIM * 2);
  _Float16* Wsp_h = (_Float16*)alloc((size_t)2 * H_DIM * H_DIM * 2);
  (void)ws_size; (void)n_in;

  auto pack = [&](const float* W, _Float16* Wp, int K, int Nc) {
    int total = (K >> 5) * (Nc >> 4) * 512;
    pack_w_kernel<<<ceil_div(total, 256), 256, 0, stream>>>(W, Wp, K, Nc);
  };
  auto zero = [&](void* p, int nwords) {
    fill0_kernel<<<ceil_div(nwords, 256), 256, 0, stream>>>((unsigned*)p, nwords);
  };

  // ---- convert + pack weights to fragment-ordered f16 once ----
  pack(Win, Win_h, IN_DIM, H_DIM);
  for (int i = 0; i < LAYERS; ++i) {
    pack(Wq  + (size_t)i * H_DIM * HC_DIM, Wq_h + (size_t)i * H_DIM * HC_DIM, H_DIM, HC_DIM);
    pack(Wk  + (size_t)i * H_DIM * HC_DIM, Wk_h + (size_t)i * H_DIM * HC_DIM, H_DIM, HC_DIM);
    pack(Wv  + (size_t)i * H_DIM * HC_DIM, Wv_h + (size_t)i * H_DIM * HC_DIM, H_DIM, HC_DIM);
    pack(Wsr + (size_t)i * H_DIM * H_DIM,  Ws_h + (size_t)i * H_DIM * H_DIM,  H_DIM, H_DIM);
    pack(W1  + (size_t)i * H_DIM * FF_DIM, W1_h + (size_t)i * H_DIM * FF_DIM, H_DIM, FF_DIM);
    pack(W2  + (size_t)i * FF_DIM * H_DIM, W2_h + (size_t)i * FF_DIM * H_DIM, FF_DIM, H_DIM);
  }
  pack(Wsp,                        Wsp_h,                        H_DIM, H_DIM);
  pack(Wsp + (size_t)H_DIM*H_DIM,  Wsp_h + (size_t)H_DIM*H_DIM,  H_DIM, H_DIM);

  const int gblk = N / 16;   // exact

  // ---- input projection: h = x @ Win + b_in ----
  wmma_gemm_kernel<IN_DIM, 0><<<gblk, 256, 0, stream>>>(x, Win_h, b_in, Hbuf, H_DIM);
  copy_kernel<<<ceil_div((long long)N * H_DIM, 256), 256, 0, stream>>>(SAV0, Hbuf, N * H_DIM);

  for (int i = 0; i < LAYERS; ++i) {
    const _Float16* Wq_i = Wq_h + (size_t)i * H_DIM * HC_DIM;
    const _Float16* Wk_i = Wk_h + (size_t)i * H_DIM * HC_DIM;
    const _Float16* Wv_i = Wv_h + (size_t)i * H_DIM * HC_DIM;
    const _Float16* Ws_i = Ws_h + (size_t)i * H_DIM * H_DIM;
    const _Float16* W1_i = W1_h + (size_t)i * H_DIM * FF_DIM;
    const _Float16* W2_i = W2_h + (size_t)i * FF_DIM * H_DIM;

    // q, k projections
    wmma_gemm_kernel<H_DIM, 0><<<gblk, 256, 0, stream>>>(
        Hbuf, Wq_i, bq + (size_t)i * HC_DIM, Qbuf, HC_DIM);
    wmma_gemm_kernel<H_DIM, 0><<<gblk, 256, 0, stream>>>(
        Hbuf, Wk_i, bk + (size_t)i * HC_DIM, Kbuf, HC_DIM);

    // alpha + segment max
    zero(MM, N * NHEAD);
    edge_alpha_kernel<<<ceil_div((long long)E * NHEAD, 256), 256, 0, stream>>>(
        Qbuf, Kbuf, srcI, dstI, ALPHA, MM, E);

    // softmax numerator + denominator
    zero(DEN, N * NHEAD);
    edge_exp_kernel<<<ceil_div((long long)E * NHEAD, 256), 256, 0, stream>>>(
        ALPHA, MM, DEN, dstI, E);

    // v projection (reuse Q buffer), aggregate into K buffer
    wmma_gemm_kernel<H_DIM, 0><<<gblk, 256, 0, stream>>>(
        Hbuf, Wv_i, bv + (size_t)i * HC_DIM, Qbuf, HC_DIM);
    zero(Kbuf, N * HC_DIM);
    edge_agg_kernel<<<ceil_div((long long)E * NHEAD * (HD_DIM / 4), 256), 256, 0, stream>>>(
        ALPHA, DEN, Qbuf, Kbuf, srcI, dstI, E);

    // root transform + head mean, then h = LN(h + a)
    wmma_gemm_kernel<H_DIM, 0><<<gblk, 256, 0, stream>>>(
        Hbuf, Ws_i, bs + (size_t)i * H_DIM, AR, H_DIM);
    head_mean_kernel<<<ceil_div((long long)N * H_DIM, 256), 256, 0, stream>>>(Kbuf, AR, N);
    ln_residual_kernel<<<ceil_div(N, 8), 256, 0, stream>>>(
        Hbuf, AR, ln1s + (size_t)i * H_DIM, ln1b + (size_t)i * H_DIM, N);

    // FFN: relu(h@W1+b1)@W2+b2, then h = LN(h + f)
    wmma_gemm_kernel<H_DIM, 1><<<gblk, 256, 0, stream>>>(
        Hbuf, W1_i, b1 + (size_t)i * FF_DIM, Qbuf, FF_DIM);
    wmma_gemm_kernel<FF_DIM, 0><<<gblk, 256, 0, stream>>>(
        Qbuf, W2_i, b2 + (size_t)i * H_DIM, AR, H_DIM);
    ln_residual_kernel<<<ceil_div(N, 8), 256, 0, stream>>>(
        Hbuf, AR, ln2s + (size_t)i * H_DIM, ln2b + (size_t)i * H_DIM, N);

    // long-range skip: h += outs[i-2] @ Wsp[p] + bsp[p]
    if (i >= 2) {
      int p = (i - 2) / 2;
      const float* sav = (i == 2) ? SAV0 : SAV1;
      wmma_gemm_kernel<H_DIM, 0><<<gblk, 256, 0, stream>>>(
          sav, Wsp_h + (size_t)p * H_DIM * H_DIM, bsp + (size_t)p * H_DIM, AR, H_DIM);
      add_kernel<<<ceil_div((long long)N * H_DIM, 256), 256, 0, stream>>>(Hbuf, AR, N * H_DIM);
    }
    if (i == 0)
      copy_kernel<<<ceil_div((long long)N * H_DIM, 256), 256, 0, stream>>>(SAV1, Hbuf, N * H_DIM);
  }

  // ---- global mean pool + output head ----
  float* out_main = (float*)d_out;          // [B]   predictions
  float* gr       = (float*)d_out + B;      // [B,H] graph_repr
  zero(POOL, B * H_DIM);
  zero(CNT, B);
  pool_cnt_kernel<<<ceil_div(N, 256), 256, 0, stream>>>(batch, CNT, N);
  pool_acc_kernel<<<ceil_div((long long)N * H_DIM, 256), 256, 0, stream>>>(Hbuf, batch, POOL, N);
  pool_final_kernel<<<ceil_div(B * H_DIM, 256), 256, 0, stream>>>(POOL, CNT, gr, B);
  head_kernel<<<1, 64, 0, stream>>>(gr, Wo1, bo1, Wo2, bo2, out_main, B);
}
